// SingleHeadAttention_43018392437332
// MI455X (gfx1250) — matrix operand order repaired and checked by reading, not compiled, not run on hardware
//
#include <hip/hip_runtime.h>

// Single-head causal attention, T=8192, C=H=1024, fp32 in/out.
// Strategy: bf16 WMMA for all matmuls (f32 accumulate), FlashAttention-style
// fused QK^T + online softmax + PV kernel. Workspace ~70 MB.

#define TT 8192
#define CC 1024
#define HH 1024

typedef unsigned short u16;
typedef __bf16 bf16x16 __attribute__((ext_vector_type(16)));
typedef float f32x8 __attribute__((ext_vector_type(8)));

union AB32 { uint4 q[2]; bf16x16 v; };

__device__ __forceinline__ u16 f2bf(float x) {
  union { float f; unsigned u; } v; v.f = x;
  unsigned r = v.u + 0x7FFFu + ((v.u >> 16) & 1u);  // round-to-nearest-even
  return (u16)(r >> 16);
}

__device__ __forceinline__ f32x8 wmma_bf16(bf16x16 a, bf16x16 b, f32x8 c) {
  return __builtin_amdgcn_wmma_f32_16x16x32_bf16(false, a, false, b, (short)0, c,
                                                 false, false);
}

// Raw LDS byte offset of a __shared__ object (for async-to-LDS VDST operand).
__device__ __forceinline__ unsigned lds_off(const void* p) {
  return (unsigned)(uintptr_t)(__attribute__((address_space(3))) const void*)p;
}

// A fragment (16x32, 16-bit): lane holds row (lo), K chunks at hi*8 and 16+hi*8.
__device__ __forceinline__ bf16x16 load_a_frag(const u16* base, int ld, int row,
                                               int k, int lo, int hi) {
  const u16* p = base + (size_t)(row + lo) * ld + k + hi * 8;
  AB32 f;
  f.q[0] = *(const uint4*)(p);
  f.q[1] = *(const uint4*)(p + 16);
  return f.v;
}

// B fragment (32x16, 16-bit): lane holds col (lo), K chunks at hi*8 and 16+hi*8.
// "base" is the [N,K]-row-major (i.e. column-contiguous) operand.
__device__ __forceinline__ bf16x16 load_b_frag(const u16* base, int ld, int n,
                                               int k, int lo, int hi) {
  const u16* p = base + (size_t)(n + lo) * ld + k + hi * 8;
  AB32 f;
  f.q[0] = *(const uint4*)(p);
  f.q[1] = *(const uint4*)(p + 16);
  return f.v;
}

// ---------------- fp32 -> bf16 convert (tokens) ----------------
__global__ void cvt_kernel(const float* __restrict__ src, u16* __restrict__ dst,
                           int n4) {
  int i = blockIdx.x * blockDim.x + threadIdx.x;
  if (i < n4) {
    float4 v = ((const float4*)src)[i];
    ushort4 o;
    o.x = f2bf(v.x); o.y = f2bf(v.y); o.z = f2bf(v.z); o.w = f2bf(v.w);
    ((ushort4*)dst)[i] = o;
  }
}

// ---------- W [C,H] fp32 -> Wt [H,C] bf16 (LDS tile transpose) ----------
__global__ void wtrans_kernel(const float* __restrict__ W, u16* __restrict__ Wt) {
  __shared__ float tile[32][33];
  int hb = blockIdx.x * 32;
  int cb = blockIdx.y * 32;
  int tx = threadIdx.x, ty = threadIdx.y;  // 32 x 8
  #pragma unroll
  for (int j = 0; j < 32; j += 8)
    tile[ty + j][tx] = W[(size_t)(cb + ty + j) * HH + hb + tx];
  __syncthreads();
  #pragma unroll
  for (int j = 0; j < 32; j += 8)
    Wt[(size_t)(hb + ty + j) * CC + cb + tx] = f2bf(tile[tx][ty + j]);
}

// ---------------- projection GEMM: Out[M,N] = A[M,K] * Bt[N,K]^T ----------------
// K = CC = 1024. 8 waves, macro tile 128x128, wave tile 32x64.
__global__ __launch_bounds__(256) void proj_kernel(const u16* __restrict__ A,
                                                   const u16* __restrict__ Bt,
                                                   u16* __restrict__ Out,
                                                   int M, int N) {
  const int tid = threadIdx.x;
  const int lane = tid & 31, wv = tid >> 5;
  const int lo = lane & 15, hi = lane >> 4;
  const int wm = wv >> 1, wn = wv & 1;
  const int row0 = blockIdx.x * 128 + wm * 32;
  const int col0 = blockIdx.y * 128 + wn * 64;

  f32x8 acc[2][4];
  #pragma unroll
  for (int mi = 0; mi < 2; ++mi)
    #pragma unroll
    for (int ni = 0; ni < 4; ++ni)
      acc[mi][ni] = f32x8{0.f, 0.f, 0.f, 0.f, 0.f, 0.f, 0.f, 0.f};

  for (int k = 0; k < CC; k += 32) {
    bf16x16 af[2], bfr[4];
    #pragma unroll
    for (int mi = 0; mi < 2; ++mi)
      af[mi] = load_a_frag(A, CC, row0 + 16 * mi, k, lo, hi);
    #pragma unroll
    for (int ni = 0; ni < 4; ++ni)
      bfr[ni] = load_b_frag(Bt, CC, col0 + 16 * ni, k, lo, hi);
    #pragma unroll
    for (int mi = 0; mi < 2; ++mi)
      #pragma unroll
      for (int ni = 0; ni < 4; ++ni)
        acc[mi][ni] = wmma_bf16(af[mi], bfr[ni], acc[mi][ni]);
  }

  #pragma unroll
  for (int mi = 0; mi < 2; ++mi)
    #pragma unroll
    for (int ni = 0; ni < 4; ++ni)
      #pragma unroll
      for (int r = 0; r < 8; ++r) {
        int row = row0 + 16 * mi + r + 8 * hi;
        int col = col0 + 16 * ni + lo;
        Out[(size_t)row * N + col] = f2bf(acc[mi][ni][r]);
      }
}

// ---------------- fused causal attention ----------------
// 512 workgroups; each owns 16 query rows. 8 waves; wave w handles S-tile
// columns [16w,16w+16) of each 128-wide key block and head slice [128w,128w+128).
__global__ __launch_bounds__(256) void attn_kernel(const u16* __restrict__ Qb,
                                                   const u16* __restrict__ Kb,
                                                   const u16* __restrict__ Vt,
                                                   float* __restrict__ out) {
  __shared__ u16 qLds[16 * 1032];    // 16 x 1024 bf16, padded (+8)
  __shared__ float sLds[16 * 132];   // 16 x 128 f32 scores, padded (+4)
  __shared__ u16 pLds[16 * 136];     // 16 x 128 bf16 probs, padded (+8)
  __shared__ float mrow[16], lrow[16], fac[16], red[16][16];

  const int tid = threadIdx.x;
  const int lane = tid & 31, wv = tid >> 5;
  const int lo = lane & 15, hi = lane >> 4;
  const int qb = (int)(gridDim.x - 1 - blockIdx.x) * 16;  // longest rows first

  // stage Q block (16 x 1024 bf16 = 32 KB) into LDS via async DMA (ASYNCcnt)
  {
    int r = tid >> 4;
    int cchunk = (tid & 15) * 64;
    unsigned lbase = lds_off(qLds) + (unsigned)(r * 1032 + cchunk) * 2u;
    const u16* g = Qb + (size_t)(qb + r) * CC + cchunk;
    #pragma unroll
    for (int i = 0; i < 8; ++i) {
      unsigned laddr = lbase + (unsigned)(i * 16);
      unsigned long long gaddr = (unsigned long long)(uintptr_t)(g + i * 8);
      asm volatile("global_load_async_to_lds_b128 %0, %1, off"
                   :: "v"(laddr), "v"(gaddr) : "memory");
    }
    asm volatile("s_wait_asynccnt 0x0" ::: "memory");
  }
  if (tid < 16) { mrow[tid] = -INFINITY; lrow[tid] = 0.f; }
  __syncthreads();

  f32x8 acc[8];
  #pragma unroll
  for (int t = 0; t < 8; ++t)
    acc[t] = f32x8{0.f, 0.f, 0.f, 0.f, 0.f, 0.f, 0.f, 0.f};

  for (int jb = 0; jb <= qb + 15; jb += 128) {
    // warm L2/L1 for the next key block while we compute on this one
    if (jb + 128 <= qb + 15) {
      __builtin_prefetch(Kb + (size_t)(jb + 128 + 16 * wv + lo) * CC + hi * 512, 0, 1);
      __builtin_prefetch(Vt + (size_t)(128 * wv + lane * 4) * TT + jb + 128, 0, 1);
    }

    // ---- S tile: Q[16x1024] . K[16x1024]^T ----
    const int jcol = jb + 16 * wv;
    if (jcol <= qb + 15) {
      f32x8 s = f32x8{0.f, 0.f, 0.f, 0.f, 0.f, 0.f, 0.f, 0.f};
      #pragma unroll 4
      for (int kc = 0; kc < CC; kc += 32) {
        AB32 af;
        const u16* ap = qLds + lo * 1032 + kc + hi * 8;
        af.q[0] = *(const uint4*)ap;
        af.q[1] = *(const uint4*)(ap + 16);
        bf16x16 bfr = load_b_frag(Kb, CC, jcol, kc, lo, hi);
        s = wmma_bf16(af.v, bfr, s);
      }
      #pragma unroll
      for (int r = 0; r < 8; ++r) {
        int i = qb + r + 8 * hi;
        int j = jcol + lo;
        float v = s[r] * 0.03125f;  // 1/sqrt(1024)
        sLds[(r + 8 * hi) * 132 + 16 * wv + lo] = (j > i) ? -INFINITY : v;
      }
    } else {
      #pragma unroll
      for (int r = 0; r < 8; ++r)
        sLds[(r + 8 * hi) * 132 + 16 * wv + lo] = -INFINITY;
    }
    __syncthreads();

    // ---- online softmax over the 16 x 128 block ----
    {
      int r = tid >> 4, c = tid & 15;
      const float* srow = sLds + r * 132 + c * 8;
      float pm = -INFINITY;
      #pragma unroll
      for (int k = 0; k < 8; ++k) pm = fmaxf(pm, srow[k]);
      red[r][c] = pm;
      __syncthreads();
      if (c == 0) {
        float bm = red[r][0];
        #pragma unroll
        for (int k = 1; k < 16; ++k) bm = fmaxf(bm, red[r][k]);
        float mo = mrow[r];
        float mn = fmaxf(mo, bm);       // finite: diag element always present
        fac[r] = __expf(mo - mn);       // mo=-inf -> 0
        mrow[r] = mn;
      }
      __syncthreads();
      float mn = mrow[r];
      float ps = 0.f;
      u16* prow = pLds + r * 136 + c * 8;
      #pragma unroll
      for (int k = 0; k < 8; ++k) {
        float e = __expf(srow[k] - mn);  // masked -inf -> 0
        ps += e;
        prow[k] = f2bf(e);
      }
      red[r][c] = ps;
      __syncthreads();
      if (c == 0) {
        float sum = 0.f;
        #pragma unroll
        for (int k = 0; k < 16; ++k) sum += red[r][k];
        lrow[r] = lrow[r] * fac[r] + sum;
      }
      __syncthreads();
    }

    // ---- rescale accumulator, then PV: O += P[16x128] . V[128 x headslice] ----
    float f8[8];
    #pragma unroll
    for (int r = 0; r < 8; ++r) f8[r] = fac[r + 8 * hi];
    #pragma unroll
    for (int t = 0; t < 8; ++t)
      #pragma unroll
      for (int r = 0; r < 8; ++r) acc[t][r] *= f8[r];

    #pragma unroll
    for (int kk = 0; kk < 128; kk += 32) {
      AB32 af;
      const u16* ap = pLds + lo * 136 + kk + hi * 8;
      af.q[0] = *(const uint4*)ap;
      af.q[1] = *(const uint4*)(ap + 16);
      #pragma unroll
      for (int t = 0; t < 8; ++t) {
        int h = 128 * wv + 16 * t;
        bf16x16 bfr = load_b_frag(Vt, TT, h, jb + kk, lo, hi);
        acc[t] = wmma_bf16(af.v, bfr, acc[t]);
      }
    }
    // next iteration's pLds writes are fenced by the barrier after S-compute;
    // every thread finished PV (program order) before passing that barrier.
  }

  float invl[8];
  #pragma unroll
  for (int r = 0; r < 8; ++r) invl[r] = 1.0f / lrow[r + 8 * hi];
  #pragma unroll
  for (int t = 0; t < 8; ++t) {
    int h0 = 128 * wv + 16 * t + lo;
    #pragma unroll
    for (int r = 0; r < 8; ++r) {
      int i = qb + r + 8 * hi;
      out[(size_t)i * HH + h0] = acc[t][r] * invl[r];
    }
  }
}

extern "C" void kernel_launch(void* const* d_in, const int* in_sizes, int n_in,
                              void* d_out, int out_size, void* d_ws, size_t ws_size,
                              hipStream_t stream) {
  const float* tokens = (const float*)d_in[0];
  const float* Wq = (const float*)d_in[1];
  const float* Wk = (const float*)d_in[2];
  const float* Wv = (const float*)d_in[3];
  // d_in[4] = training flag (0 -> dropout identity), ignored.
  float* out = (float*)d_out;

  char* ws = (char*)d_ws;
  const size_t MB = 1024 * 1024;
  u16* tok_bf = (u16*)(ws + 0);          // 16 MB  [T,C] bf16
  u16* WtQ    = (u16*)(ws + 16 * MB);    //  2 MB  [H,C] bf16
  u16* WtK    = (u16*)(ws + 18 * MB);
  u16* WtV    = (u16*)(ws + 20 * MB);
  u16* Qb     = (u16*)(ws + 22 * MB);    // 16 MB  [T,H] bf16
  u16* Kb     = (u16*)(ws + 38 * MB);    // 16 MB  [T,H] bf16
  u16* Vt     = (u16*)(ws + 54 * MB);    // 16 MB  [H,T] bf16 (transposed V)

  // 1) convert tokens to bf16
  int n4 = TT * CC / 4;
  cvt_kernel<<<(n4 + 255) / 256, 256, 0, stream>>>(tokens, tok_bf, n4);

  // 2) convert + transpose weights
  dim3 tb(32, 8);
  wtrans_kernel<<<dim3(32, 32), tb, 0, stream>>>(Wq, WtQ);
  wtrans_kernel<<<dim3(32, 32), tb, 0, stream>>>(Wk, WtK);
  wtrans_kernel<<<dim3(32, 32), tb, 0, stream>>>(Wv, WtV);

  // 3) projections: Q = X Wq, K = X Wk (row major), V^T = Wv^T X^T (row major)
  proj_kernel<<<dim3(TT / 128, HH / 128), 256, 0, stream>>>(tok_bf, WtQ, Qb, TT, HH);
  proj_kernel<<<dim3(TT / 128, HH / 128), 256, 0, stream>>>(tok_bf, WtK, Kb, TT, HH);
  proj_kernel<<<dim3(HH / 128, TT / 128), 256, 0, stream>>>(WtV, tok_bf, Vt, HH, TT);

  // 4) fused causal attention
  attn_kernel<<<TT / 16, 256, 0, stream>>>(Qb, Kb, Vt, out);
}
